// LVCBlock_1726576854965
// MI455X (gfx1250) — matrix operand order repaired and checked by reading, not compiled, not run on hardware
//
#include <hip/hip_runtime.h>
#include <hip/hip_bf16.h>
#include <math.h>

// ---------------------------------------------------------------------------
// LVCBlock for MI455X (gfx1250, wave32).  Heavy GEMMs (kernel-predictor conv
// and the location-variable-convolution einsum) run on v_wmma_f32_16x16x32_f16
// with f32 accumulation.  B panels are streamed into LDS with
// global_load_async_to_lds_b128 (ASYNCcnt path), WMMA is fed by ds_load_b128,
// and the LVC GEMM fuses the sigmoid*tanh gate through LDS, removing the 67MB
// y tensor round-trip per layer.
// ---------------------------------------------------------------------------

typedef __attribute__((ext_vector_type(16))) _Float16 v16h;
typedef __attribute__((ext_vector_type(8)))  float    v8f;

// Problem constants (match reference)
constexpr int BATCH    = 4;
constexpr int C_INCH   = 64;     // C_IN
constexpr int C_COND   = 80;
constexpr int UP       = 8;
constexpr int LAYERS   = 4;
constexpr int KS       = 3;
constexpr int HOP      = 256;
constexpr int HID      = 64;
constexpr int COND_LEN = 128;
constexpr int IN_LEN   = COND_LEN * HOP / UP;        // 4096
constexpr int OUT_LEN  = COND_LEN * HOP;             // 32768
constexpr int L_W      = C_INCH * (2 * C_INCH) * KS * LAYERS;  // 98304
constexpr int L_B      = 2 * C_INCH * LAYERS;                  // 512
constexpr int KTOT     = C_INCH * KS;                // 192 = contraction dim
constexpr int KTILES   = KTOT / 32;                  // 6
constexpr int NT_COND  = COND_LEN / 16;              // 8
constexpr int MT_KK    = L_W / 16;                   // 6144
constexpr int MT_KB    = L_B / 16;                   // 32
constexpr int MT_LVC   = (2 * C_INCH) / 16;          // 8
constexpr int NT_LVC   = HOP / 16;                   // 16
constexpr int NBLK     = COND_LEN;                   // 128 LVC blocks

// LDS budgets
constexpr int COND_B_BYTES = KTILES * NT_COND * 1024;        // 49152
constexpr int LVC_B_BYTES  = KTILES * NT_LVC * 1024;         // 98304
constexpr int SY_STRIDE    = HOP + 4;                        // pad: kill bank alias
constexpr int LVC_Y_BYTES  = 2 * C_INCH * SY_STRIDE * 4;     // 133120
constexpr int LVC_LDS      = LVC_B_BYTES + LVC_Y_BYTES;      // 231424 (< 320KB)

__device__ __forceinline__ float lrelu(float v, float s) { return v >= 0.0f ? v : s * v; }

// Async global->LDS copy of 16B per lane (CDNA5 ISA 08, GV mode).
// VDST = per-lane LDS byte offset, VADDR = 64-bit global address.
__device__ __forceinline__ void async_copy16(unsigned lds_off, const void* gsrc) {
  asm volatile("global_load_async_to_lds_b128 %0, %1, off"
               :: "v"(lds_off), "v"(gsrc) : "memory");
}
__device__ __forceinline__ void wait_async0() {
  asm volatile("s_wait_asynccnt 0x0" ::: "memory");
}

// ---------------------------------------------------------------------------
// Fragment coordinate helpers (CDNA5 ISA 7.12.2, wave32, 16-bit operands)
// A 16x32 (MxK):  M = lane&15 ; K = (h<8 ? h : h+8) + ((lane&16) ? 8 : 0)
// B 32x16 (KxN):  N = lane&15 ; K = (h<8 ? h : h+8) + ((lane&16) ? 8 : 0)
// C/D 16x16 f32:  N = lane&15 ; M = r + ((lane&16) ? 8 : 0)
// ---------------------------------------------------------------------------
__device__ __forceinline__ int a_kcoord(int lane, int h) {
  return ((h & 8) ? h + 8 : h) + ((lane & 16) ? 8 : 0);
}
__device__ __forceinline__ int b_kcoord(int lane, int h) {
  return ((h & 8) ? h + 8 : h) + ((lane & 16) ? 8 : 0);
}

// ------------------------- predictor front-end -----------------------------

// c (B,80,128) -> h0 (B,64,128), k=5 pad=2, lrelu 0.1
__global__ void k_kin(const float* __restrict__ c, const float* __restrict__ w,
                      const float* __restrict__ bias, float* __restrict__ out) {
  int idx = blockIdx.x * 256 + threadIdx.x;
  if (idx >= BATCH * HID * COND_LEN) return;
  int t  = idx & (COND_LEN - 1);
  int ch = (idx >> 7) & (HID - 1);
  int b  = idx >> 13;
  float acc = bias[ch];
  for (int ci = 0; ci < C_COND; ++ci) {
    const float* cb = c + ((size_t)b * C_COND + ci) * COND_LEN;
    const float* wr = w + (ch * C_COND + ci) * 5;
#pragma unroll
    for (int k = 0; k < 5; ++k) {
      int tt = t + k - 2;
      if (tt >= 0 && tt < COND_LEN) acc += wr[k] * cb[tt];
    }
  }
  out[idx] = lrelu(acc, 0.1f);
}

// 64->64 conv k=3 pad=1 + lrelu 0.1 over (B,64,128)
__global__ void k_res64(const float* __restrict__ in, const float* __restrict__ w,
                        const float* __restrict__ bias, float* __restrict__ out) {
  int idx = blockIdx.x * 256 + threadIdx.x;
  if (idx >= BATCH * HID * COND_LEN) return;
  int t  = idx & (COND_LEN - 1);
  int ch = (idx >> 7) & (HID - 1);
  int b  = idx >> 13;
  float acc = bias[ch];
  for (int ci = 0; ci < HID; ++ci) {
    const float* ib = in + ((size_t)b * HID + ci) * COND_LEN;
    const float* wr = w + (ch * HID + ci) * 3;
#pragma unroll
    for (int k = 0; k < 3; ++k) {
      int tt = t + k - 1;
      if (tt >= 0 && tt < COND_LEN) acc += wr[k] * ib[tt];
    }
  }
  out[idx] = lrelu(acc, 0.1f);
}

// ------------------------- fragment packing --------------------------------

// Pack row-major f32 [Mtiles*16, Ktot] weights into A fragments.
// dst layout: tile = mt*Ktiles + kt ; element = tile*512 + lane*16 + h
__global__ void k_pack_A(const float* __restrict__ src, _Float16* __restrict__ dst,
                         int Mtiles, int Ktiles, int Ktot) {
  size_t idx   = (size_t)blockIdx.x * 256 + threadIdx.x;
  size_t total = (size_t)Mtiles * Ktiles * 512;
  if (idx >= total) return;
  int    h    = idx & 15;
  int    lane = (idx >> 4) & 31;
  size_t tile = idx >> 9;
  int    kt   = (int)(tile % Ktiles);
  size_t mt   = tile / Ktiles;
  int    kg   = kt * 32 + a_kcoord(lane, h);
  size_t m    = mt * 16 + (lane & 15);
  float  v    = (kg < Ktot) ? src[m * Ktot + kg] : 0.0f;
  dst[idx] = (_Float16)v;
}

// Pack im2col of h = h0 + r (B,64,128) into B fragments for the predictor GEMM.
// tile = (b*KTILES + kt)*NT_COND + nt
__global__ void k_pack_B_cond(const float* __restrict__ h0, const float* __restrict__ r,
                              _Float16* __restrict__ dst) {
  int idx = blockIdx.x * 256 + threadIdx.x;
  if (idx >= BATCH * KTILES * NT_COND * 512) return;
  int h    = idx & 15;
  int lane = (idx >> 4) & 31;
  int tile = idx >> 9;
  int nt   = tile & (NT_COND - 1);
  int kt   = (tile >> 3) % KTILES;
  int b    = tile / (KTILES * NT_COND);
  int kg   = kt * 32 + b_kcoord(lane, h);
  int c    = kg / 3, kp = kg % 3;
  int t    = nt * 16 + (lane & 15) + kp - 1;
  float v  = 0.0f;
  if (t >= 0 && t < COND_LEN) {
    size_t o = ((size_t)b * HID + c) * COND_LEN + t;
    v = h0[o] + r[o];
  }
  dst[idx] = (_Float16)v;
}

// Gather per-(b,l) LVC A fragments (predicted kernels) for one layer.
// tile = (((b*NBLK + l)*MT_LVC + mt)*KTILES + kt)
__global__ void k_pack_lvc_A(const float* __restrict__ kkout, _Float16* __restrict__ dst,
                             int layer) {
  size_t idx = (size_t)blockIdx.x * 256 + threadIdx.x;
  if (idx >= (size_t)BATCH * NBLK * MT_LVC * KTILES * 512) return;
  int    h    = idx & 15;
  int    lane = (idx >> 4) & 31;
  size_t tile = idx >> 9;
  int    kt   = (int)(tile % KTILES);
  size_t t2   = tile / KTILES;
  int    mt   = (int)(t2 & (MT_LVC - 1));
  size_t t3   = t2 >> 3;
  int    l    = (int)(t3 & (NBLK - 1));
  int    b    = (int)(t3 >> 7);
  int    kg   = kt * 32 + a_kcoord(lane, h);
  int    ci   = kg / 3, kp = kg % 3;
  int    o    = mt * 16 + (lane & 15);
  size_t ch   = ((size_t)(layer * C_INCH + ci) * (2 * C_INCH) + o) * KS + kp;
  dst[idx] = (_Float16)kkout[((size_t)b * L_W + ch) * COND_LEN + l];
}

// Build per-(b,l) B fragments (windows of the dilated-conv output y1).
// tile = (((b*NBLK + l)*KTILES + kt)*NT_LVC + nt)
__global__ void k_pack_lvc_B(const float* __restrict__ y1, _Float16* __restrict__ dst) {
  size_t idx = (size_t)blockIdx.x * 256 + threadIdx.x;
  if (idx >= (size_t)BATCH * NBLK * KTILES * NT_LVC * 512) return;
  int    h    = idx & 15;
  int    lane = (idx >> 4) & 31;
  size_t tile = idx >> 9;
  int    nt   = (int)(tile & (NT_LVC - 1));
  size_t t2   = tile >> 4;
  int    kt   = (int)(t2 % KTILES);
  size_t t3   = t2 / KTILES;
  int    l    = (int)(t3 & (NBLK - 1));
  int    b    = (int)(t3 >> 7);
  int    kg   = kt * 32 + b_kcoord(lane, h);
  int    ci   = kg / 3, kp = kg % 3;
  long   t    = (long)l * HOP + nt * 16 + (lane & 15) + kp - 1;
  float  v    = (t >= 0 && t < OUT_LEN)
                  ? y1[((size_t)b * C_INCH + ci) * OUT_LEN + t] : 0.0f;
  dst[idx] = (_Float16)v;
}

// ------------------------- WMMA GEMMs --------------------------------------

// Predictor GEMM: 8 waves/block; per-batch B panel (48KB) async-staged into
// LDS once, wave w computes rows of tile mt = blockIdx.x*8 + w with A held in
// registers.
__global__ void k_wmma_gemm_rows(const _Float16* __restrict__ A, const _Float16* __restrict__ B,
                                 const float* __restrict__ bias, float* __restrict__ out,
                                 int Mtiles, int Ntiles, int Ntot) {
  extern __shared__ char smem[];
  _Float16* sB = (_Float16*)smem;
  int tid  = threadIdx.x;
  int wv   = tid >> 5;
  int lane = tid & 31;
  int b    = blockIdx.y;
  int mt   = blockIdx.x * 8 + wv;

  // async stage of the batch's full B panel (ASYNCcnt path)
  unsigned sbase = __builtin_amdgcn_groupstaticsize();
  {
    const uint4* gB = (const uint4*)(B + (size_t)b * KTILES * Ntiles * 512);
    int nvec = KTILES * Ntiles * 64;   // 1024B per tile = 64 x 16B
    for (int i = tid; i < nvec; i += 256)
      async_copy16(sbase + (unsigned)i * 16u, gB + i);
  }

  // A fragment loads overlap the async staging
  const v16h* Af = (const v16h*)A;
  v16h a[KTILES];
#pragma unroll
  for (int kt = 0; kt < KTILES; ++kt)
    a[kt] = Af[((size_t)mt * KTILES + kt) * 32 + lane];
  int n0   = lane & 15;
  int mrow = (lane & 16) ? 8 : 0;

  wait_async0();
  __syncthreads();

  const v16h* sBf  = (const v16h*)sB;
  float*      outb = out + (size_t)b * (size_t)Mtiles * 16 * Ntot;
  for (int nt = 0; nt < Ntiles; ++nt) {
    v8f acc = {};
#pragma unroll
    for (int kt = 0; kt < KTILES; ++kt) {
      v16h bf = sBf[(kt * Ntiles + nt) * 32 + lane];
      acc = __builtin_amdgcn_wmma_f32_16x16x32_f16(false, a[kt], false, bf,
                                                   (short)0, acc, false, false);
    }
    int col = nt * 16 + n0;
#pragma unroll
    for (int r = 0; r < 8; ++r) {
      int m = mt * 16 + mrow + r;
      outb[(size_t)m * Ntot + col] = acc[r] + bias[m];
    }
  }
}

// LVC GEMM + fused gate: per (b,l) block, 8 waves (one per 16-row M tile).
// B panel (96KB) async-staged once into LDS; per-wave 16x256 outputs (plus
// predicted bias) deposited into a padded LDS tile; after a barrier the block
// applies h += sigmoid(y[:64]) * tanh(y[64:]) directly -> no global y tensor.
__global__ void k_lvc_gemm_fused(const _Float16* __restrict__ A, const _Float16* __restrict__ B,
                                 const float* __restrict__ kbout, float* __restrict__ h,
                                 int layer) {
  extern __shared__ char smem[];
  _Float16* sB = (_Float16*)smem;
  float*    sY = (float*)(smem + LVC_B_BYTES);
  int l    = blockIdx.x;               // 128
  int b    = blockIdx.y;               // 4
  int tid  = threadIdx.x;
  int mt   = tid >> 5;                 // wave id = M tile (0..7)
  int lane = tid & 31;

  // async stage of the (b,l) B panel (ASYNCcnt path)
  unsigned sbase = __builtin_amdgcn_groupstaticsize();
  {
    const uint4* gB = (const uint4*)(B + (((size_t)b * NBLK + l) * KTILES * NT_LVC) * 512);
    for (int i = tid; i < LVC_B_BYTES / 16; i += 256)
      async_copy16(sbase + (unsigned)i * 16u, gB + i);
  }

  // per-wave A fragments + predicted bias (overlap the async staging)
  const v16h* Af = (const v16h*)A;
  v16h a[KTILES];
#pragma unroll
  for (int kt = 0; kt < KTILES; ++kt)
    a[kt] = Af[((((size_t)b * NBLK + l) * MT_LVC + mt) * KTILES + kt) * 32 + lane];
  int n0   = lane & 15;
  int mrow = (lane & 16) ? 8 : 0;
  float bv[8];
#pragma unroll
  for (int r = 0; r < 8; ++r) {
    int o = mt * 16 + mrow + r;
    bv[r] = kbout[((size_t)b * L_B + layer * (2 * C_INCH) + o) * COND_LEN + l];
  }

  wait_async0();
  __syncthreads();

  const v16h* sBf = (const v16h*)sB;
  for (int nt = 0; nt < NT_LVC; ++nt) {
    v8f acc = {};
#pragma unroll
    for (int kt = 0; kt < KTILES; ++kt) {
      v16h bf = sBf[(kt * NT_LVC + nt) * 32 + lane];
      acc = __builtin_amdgcn_wmma_f32_16x16x32_f16(false, a[kt], false, bf,
                                                   (short)0, acc, false, false);
    }
#pragma unroll
    for (int r = 0; r < 8; ++r)
      sY[(mt * 16 + mrow + r) * SY_STRIDE + nt * 16 + n0] = acc[r] + bv[r];
  }

  __syncthreads();

  // fused gate on the 64x256 output slab of this block
  float* hb = h + ((size_t)b * C_INCH) * OUT_LEN + (size_t)l * HOP;
  for (int i = tid; i < C_INCH * HOP; i += 256) {
    int o = i >> 8;            // HOP == 256
    int s = i & (HOP - 1);
    float ys = sY[o * SY_STRIDE + s];
    float yt = sY[(o + C_INCH) * SY_STRIDE + s];
    float g  = (1.0f / (1.0f + __expf(-ys))) * tanhf(yt);
    hb[(size_t)o * OUT_LEN + s] += g;
  }
}

// ------------------------- main path ---------------------------------------

// h[b,o,t] = convT(lrelu(x,0.2)); stride 8, k=16, pad 4 -> exactly 2 taps/out.
__global__ void k_convT(const float* __restrict__ x, const float* __restrict__ w,
                        const float* __restrict__ bias, float* __restrict__ h) {
  size_t idx = (size_t)blockIdx.x * 256 + threadIdx.x;
  if (idx >= (size_t)BATCH * C_INCH * OUT_LEN) return;
  int t = idx & (OUT_LEN - 1);
  int o = (idx >> 15) & (C_INCH - 1);
  int b = (int)(idx >> 21);
  int j0 = (t + 4) >> 3;
  int k0 = (t + 4) - (j0 << 3);   // in [0,8)
  int j1 = j0 - 1;
  int k1 = k0 + 8;
  float acc = bias[o];
  const float* xb = x + (size_t)b * C_INCH * IN_LEN;
  for (int ci = 0; ci < C_INCH; ++ci) {
    const float* wr = w + ((size_t)ci * C_INCH + o) * (2 * UP);
    const float* xr = xb + (size_t)ci * IN_LEN;
    if (j0 < IN_LEN) acc += wr[k0] * lrelu(xr[j0], 0.2f);
    if (j1 >= 0)     acc += wr[k1] * lrelu(xr[j1], 0.2f);
  }
  h[idx] = acc;
}

// y1 = lrelu(conv_dil(lrelu(h,0.2)), 0.2)
__global__ void k_dilconv(const float* __restrict__ h, const float* __restrict__ w,
                          const float* __restrict__ bias, float* __restrict__ y1, int dil) {
  size_t idx = (size_t)blockIdx.x * 256 + threadIdx.x;
  if (idx >= (size_t)BATCH * C_INCH * OUT_LEN) return;
  int t = idx & (OUT_LEN - 1);
  int o = (idx >> 15) & (C_INCH - 1);
  int b = (int)(idx >> 21);
  float acc = bias[o];
  const float* hb = h + (size_t)b * C_INCH * OUT_LEN;
  for (int ci = 0; ci < C_INCH; ++ci) {
    const float* wr = w + ((size_t)o * C_INCH + ci) * KS;
    const float* hr = hb + (size_t)ci * OUT_LEN;
#pragma unroll
    for (int k = 0; k < KS; ++k) {
      int tt = t + (k - 1) * dil;
      if (tt >= 0 && tt < OUT_LEN) acc += wr[k] * lrelu(hr[tt], 0.2f);
    }
  }
  y1[idx] = lrelu(acc, 0.2f);
}

// ---------------------------------------------------------------------------

static inline int gridFor(size_t n) { return (int)((n + 255) / 256); }

extern "C" void kernel_launch(void* const* d_in, const int* in_sizes, int n_in,
                              void* d_out, int out_size, void* d_ws, size_t ws_size,
                              hipStream_t stream) {
  const float* x      = (const float*)d_in[0];
  const float* c      = (const float*)d_in[1];
  const float* up_w   = (const float*)d_in[2];
  const float* up_b   = (const float*)d_in[3];
  const float* kin_w  = (const float*)d_in[4];
  const float* kin_b  = (const float*)d_in[5];
  const float* kres_w = (const float*)d_in[6];
  const float* kres_b = (const float*)d_in[7];
  const float* kk_w   = (const float*)d_in[8];
  const float* kk_b   = (const float*)d_in[9];
  const float* kb_w   = (const float*)d_in[10];
  const float* kb_b   = (const float*)d_in[11];
  const float* conv_w = (const float*)d_in[12];
  const float* conv_b = (const float*)d_in[13];
  float* h = (float*)d_out;   // (B,64,32768) residual stream lives in d_out

  // Workspace carve
  char* p = (char*)d_ws;
  auto carve = [&](size_t bytes) -> void* {
    void* r = (void*)p;
    p += (bytes + 255) & ~(size_t)255;
    return r;
  };
  const size_t condElems = (size_t)BATCH * HID * COND_LEN;
  float*    h0    = (float*)carve(condElems * 4);
  float*    ra    = (float*)carve(condElems * 4);
  float*    rb    = (float*)carve(condElems * 4);
  _Float16* condB = (_Float16*)carve((size_t)BATCH * KTILES * NT_COND * 512 * 2);
  _Float16* Akk   = (_Float16*)carve((size_t)MT_KK * KTILES * 512 * 2);
  _Float16* Akb   = (_Float16*)carve((size_t)MT_KB * KTILES * 512 * 2);
  float*    kkout = (float*)carve((size_t)BATCH * L_W * COND_LEN * 4);
  float*    kbout = (float*)carve((size_t)BATCH * L_B * COND_LEN * 4);
  float*    y1    = (float*)carve((size_t)BATCH * C_INCH * OUT_LEN * 4);
  _Float16* lvcA  = (_Float16*)carve((size_t)BATCH * NBLK * MT_LVC * KTILES * 512 * 2);
  _Float16* lvcB  = (_Float16*)carve((size_t)BATCH * NBLK * KTILES * NT_LVC * 512 * 2);

  // --- kernel predictor front-end ---
  k_kin<<<gridFor(condElems), 256, 0, stream>>>(c, kin_w, kin_b, h0);
  const float* rin = h0;
  float* rout = ra;
  for (int j = 0; j < 6; ++j) {
    k_res64<<<gridFor(condElems), 256, 0, stream>>>(
        rin, kres_w + (size_t)j * HID * HID * KS, kres_b + (size_t)j * HID, rout);
    rin  = rout;
    rout = (rout == ra) ? rb : ra;
  }
  const float* rfin = rin;  // final residual output

  // --- pack operands, run predictor GEMMs (WMMA, async-LDS-staged B) ---
  k_pack_B_cond<<<gridFor((size_t)BATCH * KTILES * NT_COND * 512), 256, 0, stream>>>(
      h0, rfin, condB);
  k_pack_A<<<gridFor((size_t)MT_KK * KTILES * 512), 256, 0, stream>>>(
      kk_w, Akk, MT_KK, KTILES, KTOT);
  k_pack_A<<<gridFor((size_t)MT_KB * KTILES * 512), 256, 0, stream>>>(
      kb_w, Akb, MT_KB, KTILES, KTOT);
  k_wmma_gemm_rows<<<dim3(MT_KK / 8, BATCH), 256, COND_B_BYTES, stream>>>(
      Akk, condB, kk_b, kkout, MT_KK, NT_COND, COND_LEN);
  k_wmma_gemm_rows<<<dim3(MT_KB / 8, BATCH), 256, COND_B_BYTES, stream>>>(
      Akb, condB, kb_b, kbout, MT_KB, NT_COND, COND_LEN);

  // --- upsample into residual stream ---
  const size_t mainElems = (size_t)BATCH * C_INCH * OUT_LEN;
  k_convT<<<gridFor(mainElems), 256, 0, stream>>>(x, up_w, up_b, h);

  // --- 4 LVC layers (WMMA + fused gate) ---
  int dil = 1;
  for (int layer = 0; layer < LAYERS; ++layer) {
    k_dilconv<<<gridFor(mainElems), 256, 0, stream>>>(
        h, conv_w + (size_t)layer * C_INCH * C_INCH * KS, conv_b + (size_t)layer * C_INCH,
        y1, dil);
    k_pack_lvc_A<<<gridFor((size_t)BATCH * NBLK * MT_LVC * KTILES * 512), 256, 0, stream>>>(
        kkout, lvcA, layer);
    k_pack_lvc_B<<<gridFor((size_t)BATCH * NBLK * KTILES * NT_LVC * 512), 256, 0, stream>>>(
        y1, lvcB);
    k_lvc_gemm_fused<<<dim3(NBLK, BATCH), 256, LVC_LDS, stream>>>(
        lvcA, lvcB, kbout, h, layer);
    dil *= 3;
  }
  (void)n_in; (void)in_sizes; (void)out_size; (void)ws_size;
}
